// LFLSTM_78950088835618
// MI455X (gfx1250) — compile-verified, matrix operand order checked by reading, not run on hardware
//
#include <hip/hip_runtime.h>
#include <hip/hip_bf16.h>

typedef __bf16 bf16_t;
typedef __attribute__((ext_vector_type(8)))  __bf16 v8bf;
typedef __attribute__((ext_vector_type(16))) __bf16 v16bf;
typedef __attribute__((ext_vector_type(8)))  float  v8f;

#define BATCH 128
#define SEQ   256
#define HCATN 1024

// ---------- helpers ----------

__device__ __forceinline__ bf16_t f2bf(float f) {
    union { float f; unsigned u; } v; v.f = f;
    unsigned r = v.u + 0x7fffu + ((v.u >> 16) & 1u);   // round-to-nearest-even
    union { unsigned short u; bf16_t b; } o;
    o.u = (unsigned short)(r >> 16);
    return o.b;
}

__device__ __forceinline__ float sigf(float x) {
    return 1.0f / (1.0f + __expf(-x));
}

__device__ __forceinline__ v16bf cat8(v8bf lo, v8bf hi) {
    v16bf r;
#pragma unroll
    for (int i = 0; i < 8; ++i) { r[i] = lo[i]; r[i + 8] = hi[i]; }
    return r;
}

// Load one 16x32 bf16 WMMA A/B fragment.
//  A (16x32, MxK): lane L holds row m=L%16; K chunks: base=(L/16)*8, elems
//  [base..base+7] and [base+16..base+23]  -> two 16-byte loads.
//  B (32x16, KxN): identical pattern with "row" = output column = weight row.
__device__ __forceinline__ v16bf load_frag(const bf16_t* base, int ld, int lane) {
    const int m  = lane & 15;
    const int ko = (lane >> 4) << 3;
    const bf16_t* p = base + (size_t)m * ld + ko;
    v8bf lo = *reinterpret_cast<const v8bf*>(p);
    v8bf hi = *reinterpret_cast<const v8bf*>(p + 16);
    return cat8(lo, hi);
}

// ---------- conversion kernels ----------

__global__ void cvt_pad_kernel(const float* __restrict__ src, bf16_t* __restrict__ dst,
                               int rows, int cols, int colsPad) {
    int idx = blockIdx.x * 256 + threadIdx.x;
    int total = rows * colsPad;
    if (idx >= total) return;
    int r = idx / colsPad, c = idx - r * colsPad;
    float v = (c < cols) ? src[(size_t)r * cols + c] : 0.0f;
    dst[idx] = f2bf(v);
}

__global__ void bias_sum_kernel(const float* __restrict__ a, const float* __restrict__ b,
                                float* __restrict__ o, int n) {
    int i = blockIdx.x * 256 + threadIdx.x;
    if (i < n) o[i] = a[i] + b[i];
}

// ---------- fused LSTM recurrence ----------
// One workgroup (512 threads, 16 waves) = 16 batch rows of one LSTM. Each wave
// owns HID/16 hidden columns and computes the i/f/g/o tiles for those columns,
// so the cell update is wave-local. h is double-buffered in LDS (bf16): step t
// reads buf[cur], writes buf[cur^1] -> one barrier per timestep. c-state stays
// in VGPRs. A/B fragments are software-pipelined (load k+1 while WMMA'ing k)
// so WMMAs don't serialize on the L2 load latency.

template <int HID, int DPAD>
__device__ void lstm_rec_impl(const bf16_t* __restrict__ xb,    // [B*T][DPAD]
                              const bf16_t* __restrict__ wih,   // [4*HID][DPAD]
                              const bf16_t* __restrict__ whh,   // [4*HID][HID]
                              const float*  __restrict__ bias,  // [4*HID]
                              bf16_t* __restrict__ hcat,        // [B*T][1024]
                              int hoff, int btile, bf16_t* hbuf /* 2*16*HID */) {
    constexpr int WAVES = 16;
    constexpr int CPW   = HID / WAVES;   // hidden cols per wave
    constexpr int HGRP  = CPW / 16;      // 16x16 tiles per gate per wave
    constexpr int KX    = DPAD / 32;
    constexpr int KH    = HID / 32;
    constexpr int HBUF  = 16 * HID;

    const int tid  = threadIdx.x;
    const int wave = tid >> 5;
    const int lane = tid & 31;
    const int rb   = btile * 16;
    const int n    = lane & 15;

    // preload per-lane biases (gate order i,f,g,o along the 4*HID axis)
    float bv[HGRP][4];
#pragma unroll
    for (int ht = 0; ht < HGRP; ++ht)
#pragma unroll
        for (int g = 0; g < 4; ++g)
            bv[ht][g] = bias[g * HID + wave * CPW + ht * 16 + n];

    for (int i = tid; i < HBUF; i += 512) hbuf[i] = f2bf(0.0f);
    __syncthreads();

    v8f cst[HGRP];
#pragma unroll
    for (int ht = 0; ht < HGRP; ++ht) { v8f z = {}; cst[ht] = z; }

    int cur = 0;
    for (int t = 0; t < SEQ; ++t) {
        const bf16_t* __restrict__ hread  = hbuf + cur * HBUF;
        bf16_t* __restrict__       hwrite = hbuf + (cur ^ 1) * HBUF;
        const bf16_t* xbase = xb + ((size_t)rb * SEQ + t) * DPAD;

        // warm L2/L0 with next timestep's x rows (the only new global data)
        if (t + 1 < SEQ)
            __builtin_prefetch(xbase + (size_t)DPAD + (lane & 15) * (SEQ * DPAD), 0, 1);

#pragma unroll
        for (int ht = 0; ht < HGRP; ++ht) {
            const int colbase = wave * CPW + ht * 16;
            v8f acc[4];
#pragma unroll
            for (int g = 0; g < 4; ++g)
#pragma unroll
                for (int r = 0; r < 8; ++r) acc[g][r] = bv[ht][g];

            // ---- x * W_ih^T (A from global bf16 x at time t), pipelined ----
            {
                v16bf a0 = load_frag(xbase, SEQ * DPAD, lane);
                v16bf b0[4];
#pragma unroll
                for (int g = 0; g < 4; ++g)
                    b0[g] = load_frag(wih + (size_t)(g * HID + colbase) * DPAD, DPAD, lane);
                for (int kb = 0; kb < KX; ++kb) {
                    v16bf a1 = a0;
                    v16bf b1[4];
#pragma unroll
                    for (int g = 0; g < 4; ++g) b1[g] = b0[g];
                    if (kb + 1 < KX) {
                        a1 = load_frag(xbase + (kb + 1) * 32, SEQ * DPAD, lane);
#pragma unroll
                        for (int g = 0; g < 4; ++g)
                            b1[g] = load_frag(
                                wih + (size_t)(g * HID + colbase) * DPAD + (kb + 1) * 32,
                                DPAD, lane);
                    }
#pragma unroll
                    for (int g = 0; g < 4; ++g)
                        acc[g] = __builtin_amdgcn_wmma_f32_16x16x32_bf16(
                            false, a0, false, b0[g], (short)0, acc[g], false, false);
                    a0 = a1;
#pragma unroll
                    for (int g = 0; g < 4; ++g) b0[g] = b1[g];
                }
            }

            // ---- h_prev * W_hh^T (A from LDS), pipelined ----
            {
                v16bf a0 = load_frag(hread, HID, lane);
                v16bf b0[4];
#pragma unroll
                for (int g = 0; g < 4; ++g)
                    b0[g] = load_frag(whh + (size_t)(g * HID + colbase) * HID, HID, lane);
                for (int kb = 0; kb < KH; ++kb) {
                    v16bf a1 = a0;
                    v16bf b1[4];
#pragma unroll
                    for (int g = 0; g < 4; ++g) b1[g] = b0[g];
                    if (kb + 1 < KH) {
                        a1 = load_frag(hread + (kb + 1) * 32, HID, lane);
#pragma unroll
                        for (int g = 0; g < 4; ++g)
                            b1[g] = load_frag(
                                whh + (size_t)(g * HID + colbase) * HID + (kb + 1) * 32,
                                HID, lane);
                    }
#pragma unroll
                    for (int g = 0; g < 4; ++g)
                        acc[g] = __builtin_amdgcn_wmma_f32_16x16x32_bf16(
                            false, a0, false, b0[g], (short)0, acc[g], false, false);
                    a0 = a1;
#pragma unroll
                    for (int g = 0; g < 4; ++g) b0[g] = b1[g];
                }
            }

            // wave-local cell update for this 16-column tile
            const int col = colbase + n;
#pragma unroll
            for (int r = 0; r < 8; ++r) {
                const int m = r + ((lane >> 4) << 3);
                float iv = acc[0][r];
                float fv = acc[1][r];
                float gv = acc[2][r];
                float ov = acc[3][r];
                float c  = sigf(fv) * cst[ht][r] + sigf(iv) * tanhf(gv);
                float h  = sigf(ov) * tanhf(c);
                cst[ht][r] = c;
                bf16_t hb = f2bf(h);
                hwrite[m * HID + col] = hb;
                hcat[((size_t)(rb + m) * SEQ + t) * HCATN + hoff + col] = hb;
            }
        }
        __syncthreads();
        cur ^= 1;
    }
}

__global__ __launch_bounds__(512) void lstm_rec_all_kernel(
    const bf16_t* xb0, const bf16_t* xb1, const bf16_t* xb2,
    const bf16_t* wih0, const bf16_t* wih1, const bf16_t* wih2,
    const bf16_t* whh0, const bf16_t* whh1, const bf16_t* whh2,
    const float* bias0, const float* bias1, const float* bias2,
    bf16_t* hcat) {
    __shared__ __attribute__((aligned(16))) bf16_t hbuf[2 * 16 * 512];
    int bid = blockIdx.x;
    if (bid < 8)
        lstm_rec_impl<512, 320>(xb0, wih0, whh0, bias0, hcat, 0,   bid,      hbuf);
    else if (bid < 16)
        lstm_rec_impl<256, 96>(xb1, wih1, whh1, bias1, hcat, 512, bid - 8,  hbuf);
    else
        lstm_rec_impl<256, 64>(xb2, wih2, whh2, bias2, hcat, 768, bid - 16, hbuf);
}

// ---------- head: FC1 (WMMA) + ReLU ----------

__global__ __launch_bounds__(256) void fc1_kernel(const bf16_t* __restrict__ hcat, // [32768][1024]
                                                  const bf16_t* __restrict__ w1,   // [256][1024]
                                                  const float*  __restrict__ b1,
                                                  float* __restrict__ z) {         // [32768][256]
    const int wave = threadIdx.x >> 5, lane = threadIdx.x & 31;
    const int tile = blockIdx.x * 8 + wave;        // 2048 m-tiles * 16 n-tiles
    const int mt = tile >> 4, nt = tile & 15;
    const int rb = mt * 16, cb = nt * 16;
    const int n = lane & 15;

    v8f acc;
    float bvv = b1[cb + n];
#pragma unroll
    for (int r = 0; r < 8; ++r) acc[r] = bvv;

    constexpr int KT = 1024 / 32;
    v16bf a0 = load_frag(hcat + (size_t)rb * 1024, 1024, lane);
    v16bf b0 = load_frag(w1   + (size_t)cb * 1024, 1024, lane);
    for (int kb = 0; kb < KT; ++kb) {
        v16bf a1 = a0, b1f = b0;
        if (kb + 1 < KT) {
            a1  = load_frag(hcat + (size_t)rb * 1024 + (kb + 1) * 32, 1024, lane);
            b1f = load_frag(w1   + (size_t)cb * 1024 + (kb + 1) * 32, 1024, lane);
        }
        acc = __builtin_amdgcn_wmma_f32_16x16x32_bf16(false, a0, false, b0,
                                                      (short)0, acc, false, false);
        a0 = a1; b0 = b1f;
    }
#pragma unroll
    for (int r = 0; r < 8; ++r) {
        const int m = r + ((lane >> 4) << 3);
        float v = acc[r];
        z[(size_t)(rb + m) * 256 + cb + n] = v > 0.0f ? v : 0.0f;
    }
}

// ---------- head: FC2 + log_softmax ----------

__global__ __launch_bounds__(256) void fc2_lsm_kernel(const float* __restrict__ z,
                                                      const float* __restrict__ w2, // [7][256]
                                                      const float* __restrict__ b2,
                                                      float* __restrict__ out) {    // [32768][7]
    __shared__ float w2s[7 * 256];
    __shared__ float b2s[7];
    for (int i = threadIdx.x; i < 7 * 256; i += 256) w2s[i] = w2[i];
    if (threadIdx.x < 7) b2s[threadIdx.x] = b2[threadIdx.x];
    __syncthreads();

    const int row = blockIdx.x * 256 + threadIdx.x;
    float s[7];
#pragma unroll
    for (int o = 0; o < 7; ++o) s[o] = b2s[o];
    const float* zr = z + (size_t)row * 256;
    for (int c = 0; c < 256; ++c) {
        float zv = zr[c];
#pragma unroll
        for (int o = 0; o < 7; ++o) s[o] += zv * w2s[o * 256 + c];
    }
    float mx = s[0];
#pragma unroll
    for (int o = 1; o < 7; ++o) mx = fmaxf(mx, s[o]);
    float se = 0.0f;
#pragma unroll
    for (int o = 0; o < 7; ++o) se += __expf(s[o] - mx);
    float lse = __logf(se);
#pragma unroll
    for (int o = 0; o < 7; ++o) out[(size_t)row * 7 + o] = s[o] - mx - lse;
}

// ---------- host-side launch ----------

extern "C" void kernel_launch(void* const* d_in, const int* in_sizes, int n_in,
                              void* d_out, int out_size, void* d_ws, size_t ws_size,
                              hipStream_t stream) {
    (void)in_sizes; (void)n_in; (void)out_size; (void)ws_size;

    const float* x0    = (const float*)d_in[0];
    const float* x1    = (const float*)d_in[1];
    const float* x2    = (const float*)d_in[2];
    const float* Wih0  = (const float*)d_in[3];
    const float* Whh0  = (const float*)d_in[4];
    const float* bih0  = (const float*)d_in[5];
    const float* bhh0  = (const float*)d_in[6];
    const float* Wih1  = (const float*)d_in[7];
    const float* Whh1  = (const float*)d_in[8];
    const float* bih1  = (const float*)d_in[9];
    const float* bhh1  = (const float*)d_in[10];
    const float* Wih2  = (const float*)d_in[11];
    const float* Whh2  = (const float*)d_in[12];
    const float* bih2  = (const float*)d_in[13];
    const float* bhh2  = (const float*)d_in[14];
    const float* fcW1  = (const float*)d_in[15];
    const float* fcb1  = (const float*)d_in[16];
    const float* fcW2  = (const float*)d_in[17];
    const float* fcb2  = (const float*)d_in[18];

    char* ws = (char*)d_ws;
    size_t off = 0;
    auto alloc = [&](size_t bytes) -> void* {
        void* p = ws + off;
        off += (bytes + 255) & ~(size_t)255;
        return p;
    };

    const int BT = BATCH * SEQ;   // 32768

    bf16_t* x0b   = (bf16_t*)alloc((size_t)BT * 320 * 2);
    bf16_t* x1b   = (bf16_t*)alloc((size_t)BT * 96  * 2);
    bf16_t* x2b   = (bf16_t*)alloc((size_t)BT * 64  * 2);
    bf16_t* wih0b = (bf16_t*)alloc((size_t)2048 * 320 * 2);
    bf16_t* wih1b = (bf16_t*)alloc((size_t)1024 * 96  * 2);
    bf16_t* wih2b = (bf16_t*)alloc((size_t)1024 * 64  * 2);
    bf16_t* whh0b = (bf16_t*)alloc((size_t)2048 * 512 * 2);
    bf16_t* whh1b = (bf16_t*)alloc((size_t)1024 * 256 * 2);
    bf16_t* whh2b = (bf16_t*)alloc((size_t)1024 * 256 * 2);
    float*  bias0 = (float*)alloc((size_t)2048 * 4);
    float*  bias1 = (float*)alloc((size_t)1024 * 4);
    float*  bias2 = (float*)alloc((size_t)1024 * 4);
    bf16_t* w1b   = (bf16_t*)alloc((size_t)256 * 1024 * 2);
    bf16_t* hcatb = (bf16_t*)alloc((size_t)BT * HCATN * 2);
    float*  zbuf  = (float*)alloc((size_t)BT * 256 * 4);

    auto cvt = [&](const float* s, bf16_t* d, int rows, int cols, int colsPad) {
        int total = rows * colsPad;
        cvt_pad_kernel<<<(total + 255) / 256, 256, 0, stream>>>(s, d, rows, cols, colsPad);
    };

    cvt(x0, x0b, BT, 300, 320);
    cvt(x1, x1b, BT, 74,  96);
    cvt(x2, x2b, BT, 35,  64);
    cvt(Wih0, wih0b, 2048, 300, 320);
    cvt(Wih1, wih1b, 1024, 74,  96);
    cvt(Wih2, wih2b, 1024, 35,  64);
    cvt(Whh0, whh0b, 2048, 512, 512);
    cvt(Whh1, whh1b, 1024, 256, 256);
    cvt(Whh2, whh2b, 1024, 256, 256);
    cvt(fcW1, w1b, 256, 1024, 1024);

    bias_sum_kernel<<<(2048 + 255) / 256, 256, 0, stream>>>(bih0, bhh0, bias0, 2048);
    bias_sum_kernel<<<(1024 + 255) / 256, 256, 0, stream>>>(bih1, bhh1, bias1, 1024);
    bias_sum_kernel<<<(1024 + 255) / 256, 256, 0, stream>>>(bih2, bhh2, bias2, 1024);

    lstm_rec_all_kernel<<<24, 512, 0, stream>>>(x0b, x1b, x2b,
                                                wih0b, wih1b, wih2b,
                                                whh0b, whh1b, whh2b,
                                                bias0, bias1, bias2, hcatb);

    fc1_kernel<<<(BT / 16) * 16 / 8, 256, 0, stream>>>(hcatb, w1b, fcb1, zbuf);

    fc2_lsm_kernel<<<BT / 256, 256, 0, stream>>>(zbuf, fcW2, fcb2, (float*)d_out);
}
